// BEVGNNModel_81492709474595
// MI455X (gfx1250) — compile-verified, matrix operand order checked by reading, not run on hardware
//
#include <hip/hip_runtime.h>

// ---------------------------------------------------------------------------
// Types for CDNA5 WMMA
// ---------------------------------------------------------------------------
typedef __attribute__((ext_vector_type(16))) __bf16 v16bf;
typedef __attribute__((ext_vector_type(8)))  float  v8f;
typedef __attribute__((ext_vector_type(4)))  unsigned int v4u;
typedef __attribute__((ext_vector_type(4)))  int    v4i;

union BFrag { v16bf v; v4u q[2]; unsigned short u[16]; };

#if __has_builtin(__builtin_amdgcn_global_load_async_to_lds_b128) && \
    __has_builtin(__builtin_amdgcn_s_wait_asynccnt)
#define HAVE_ASYNC_LDS 1
// Parameter types per hipcc diagnostic: AS1 (global) int4*, AS3 (LDS) int4*
typedef __attribute__((address_space(1))) v4i* gv4i_p;
typedef __attribute__((address_space(3))) v4i* lv4i_p;
#endif

// ---------------------------------------------------------------------------
// Problem constants (from reference)
// ---------------------------------------------------------------------------
#define E_TOT  4096
#define NNODE  1024
#define HEADS  12
#define ECH    256          // edge chunk (16 chunks)

// GEMM tiling: block = 256 threads = 8 waves; wave owns 32(M) x 64(N)
#define BM 256
#define BN 64
#define BK 32
#define LDA 40              // halves; 80B stride: 16B-aligned + conflict-free
#define LDB 40

static inline int ceil32(int x) { return (x + 31) & ~31; }
static inline int ceil64(int x) { return (x + 63) & ~63; }

// ---------------------------------------------------------------------------
// Helpers
// ---------------------------------------------------------------------------
__device__ __forceinline__ unsigned short f2bf(float f) {
    unsigned u = __float_as_uint(f);
    unsigned r = u + 0x7FFFu + ((u >> 16) & 1u);   // round-to-nearest-even
    return (unsigned short)(r >> 16);
}

__device__ __forceinline__ float gelu_tanh(float x) {
    float t = tanhf(0.7978845608028654f * (x + 0.044715f * x * x * x));
    return 0.5f * x * (1.0f + t);
}

// 16B-aligned LDS fragment load: 8 contiguous halves at base, 8 at base+16
__device__ __forceinline__ v16bf ldfrag(const unsigned short* base) {
    BFrag f;
    f.q[0] = *(const v4u*)(base);
    f.q[1] = *(const v4u*)(base + 16);
    return f.v;
}

#define WMMA_BF16(a, b, c) __builtin_amdgcn_wmma_f32_16x16x32_bf16( \
        false, (a), false, (b), (short)0, (c), false, false)

// ---------------------------------------------------------------------------
// Generic WMMA GEMM:  C[M,N] = act(A * Bt^T + bias) + resid
//   A  : bf16, [M][Kp] (K zero-padded to Kp, multiple of 32)
//   Bt : bf16 weights, PRE-TRANSPOSED [>=N rows][Kp], K-pad zeroed
//   C  : f32 (cbf16=0) or bf16 (cbf16=1), dense stride N; bias/resid f32
//   act: 0 = none, 1 = gelu(tanh)
//   M multiple of BM. No K/N guards in staging: pads are zero / masked.
// ---------------------------------------------------------------------------
__global__ __launch_bounds__(256)
void gemm_bf16_wmma(const unsigned short* __restrict__ A,
                    const unsigned short* __restrict__ Bt,
                    const float* __restrict__ bias, const float* __restrict__ resid,
                    void* __restrict__ Cv, int M, int Kp, int N, int act, int cbf16)
{
    __shared__ __align__(16) unsigned short As[2][BM * LDA];
    __shared__ __align__(16) unsigned short Bs[2][BN * LDB];   // [n][k]

    const int bm   = blockIdx.x * BM;
    const int bn   = blockIdx.y * BN;
    const int tid  = threadIdx.x;
    const int lane = tid & 31;
    const int wave = tid >> 5;                 // 0..7 -> 32 output rows each
    const int half = lane >> 4;                // 0/1
    const int r    = lane & 15;

    v8f acc[2][4] = {};                        // 2 M-subtiles x 4 N-subtiles

    const int nk = Kp / BK;

    // A tile: 256 rows x 32 halves; 4 groups of 8 halves (16B) per thread
    const int a_row0 = tid >> 2;               // + it*64
    const int a_kk   = (tid & 3) << 3;         // 0,8,16,24
    // B tile: 64 rows(n) x 32 halves(k); 1 group of 8 halves per thread
    const int b_n  = tid >> 2;
    const int b_kk = (tid & 3) << 3;

    // preload all six fragments, then 8 back-to-back WMMAs
    auto compute = [&](int cur) {
        const unsigned short* aP = As[cur];
        const unsigned short* bP = Bs[cur];
        v16bf af0 = ldfrag(&aP[(wave * 32 +      r) * LDA + (half << 3)]);
        v16bf af1 = ldfrag(&aP[(wave * 32 + 16 + r) * LDA + (half << 3)]);
        v16bf bf0 = ldfrag(&bP[( 0 + r) * LDB + (half << 3)]);
        v16bf bf1 = ldfrag(&bP[(16 + r) * LDB + (half << 3)]);
        v16bf bf2 = ldfrag(&bP[(32 + r) * LDB + (half << 3)]);
        v16bf bf3 = ldfrag(&bP[(48 + r) * LDB + (half << 3)]);
        acc[0][0] = WMMA_BF16(af0, bf0, acc[0][0]);
        acc[1][0] = WMMA_BF16(af1, bf0, acc[1][0]);
        acc[0][1] = WMMA_BF16(af0, bf1, acc[0][1]);
        acc[1][1] = WMMA_BF16(af1, bf1, acc[1][1]);
        acc[0][2] = WMMA_BF16(af0, bf2, acc[0][2]);
        acc[1][2] = WMMA_BF16(af1, bf2, acc[1][2]);
        acc[0][3] = WMMA_BF16(af0, bf3, acc[0][3]);
        acc[1][3] = WMMA_BF16(af1, bf3, acc[1][3]);
    };

#ifdef HAVE_ASYNC_LDS
    // ---- CDNA5 async global->LDS staging (ASYNCcnt), no guards ----
    auto issueAsync = [&](int k0, int buf) {
        #pragma unroll
        for (int it = 0; it < 4; ++it) {
            int row = a_row0 + it * 64;
            __builtin_amdgcn_global_load_async_to_lds_b128(
                (gv4i_p)&A[(size_t)(bm + row) * Kp + k0 + a_kk],
                (lv4i_p)&As[buf][row * LDA + a_kk], 0, 0);
        }
        __builtin_amdgcn_global_load_async_to_lds_b128(
            (gv4i_p)&Bt[(size_t)(bn + b_n) * Kp + k0 + b_kk],
            (lv4i_p)&Bs[buf][b_n * LDB + b_kk], 0, 0);
    };

    issueAsync(0, 0);
    for (int kt = 0; kt < nk; ++kt) {
        const int cur = kt & 1;
        __builtin_amdgcn_s_wait_asynccnt(0);       // tile kt landed in LDS
        __syncthreads();                           // all waves past compute kt-1
        if (kt + 1 < nk) issueAsync((kt + 1) * BK, 1 - cur);
        compute(cur);
    }
#else
    // ---- fallback: register staging (global b128 -> ds_store_b128) ----
    uint4 ra[4];
    uint4 rb;
    auto loadG = [&](int k0) {
        #pragma unroll
        for (int it = 0; it < 4; ++it)
            ra[it] = *(const uint4*)&A[(size_t)(bm + a_row0 + it * 64) * Kp + k0 + a_kk];
        rb = *(const uint4*)&Bt[(size_t)(bn + b_n) * Kp + k0 + b_kk];
    };
    auto storeL = [&](int buf) {
        #pragma unroll
        for (int it = 0; it < 4; ++it)
            *(uint4*)&As[buf][(a_row0 + it * 64) * LDA + a_kk] = ra[it];
        *(uint4*)&Bs[buf][b_n * LDB + b_kk] = rb;
    };

    loadG(0);
    storeL(0);
    __syncthreads();
    for (int kt = 0; kt < nk; ++kt) {
        const int cur = kt & 1;
        if (kt + 1 < nk) loadG((kt + 1) * BK);
        compute(cur);
        if (kt + 1 < nk) storeL(1 - cur);
        __syncthreads();
    }
#endif

    // ---- epilogue: C VGPR v -> rows M = v + 8*half within each 16x16 tile ----
    #pragma unroll
    for (int mt = 0; mt < 2; ++mt) {
        const int row0 = bm + wave * 32 + mt * 16 + half * 8;
        #pragma unroll
        for (int nt = 0; nt < 4; ++nt) {
            int col = bn + nt * 16 + r;
            if (col >= N) continue;
            float bv = bias ? bias[col] : 0.0f;
            #pragma unroll
            for (int v = 0; v < 8; ++v) {
                int row = row0 + v;
                float c = acc[mt][nt][v] + bv;
                if (act == 1) c = gelu_tanh(c);
                if (resid) c += resid[(size_t)row * N + col];
                if (cbf16) ((unsigned short*)Cv)[(size_t)row * N + col] = f2bf(c);
                else       ((float*)Cv)[(size_t)row * N + col] = c;
            }
        }
    }
}

// ---------------------------------------------------------------------------
// Weight convert + transpose:  W[K][N] f32  ->  Wt[N][Kp] bf16, K-pad zeroed
// ---------------------------------------------------------------------------
__global__ __launch_bounds__(256)
void wconv_kernel(const float* __restrict__ W, unsigned short* __restrict__ Wt,
                  int K, int Kp, int N)
{
    size_t i = (size_t)blockIdx.x * blockDim.x + threadIdx.x;
    size_t tot = (size_t)N * Kp;
    if (i >= tot) return;
    int n = (int)(i / Kp);
    int k = (int)(i % Kp);
    Wt[i] = (k < K) ? f2bf(W[(size_t)k * N + n]) : (unsigned short)0;
}

// ---------------------------------------------------------------------------
// LayerNorm over last dim D: f32 in (stride D)
//   obf16=1: bf16 out, stride Dp, pad cols zeroed;  obf16=0: f32 out, stride D
// ---------------------------------------------------------------------------
__global__ __launch_bounds__(128)
void ln_kernel(const float* __restrict__ X, const float* __restrict__ g,
               const float* __restrict__ b, void* __restrict__ Yv,
               int M, int D, int Dp, int obf16)
{
    int row  = blockIdx.x * 4 + (threadIdx.x >> 5);
    int lane = threadIdx.x & 31;
    if (row >= M) return;
    const float* x = X + (size_t)row * D;
    float s = 0.f, s2 = 0.f;
    for (int i = lane; i < D; i += 32) { float v = x[i]; s += v; s2 += v * v; }
    #pragma unroll
    for (int off = 16; off; off >>= 1) {
        s  += __shfl_xor(s,  off, 32);
        s2 += __shfl_xor(s2, off, 32);
    }
    float mean = s / (float)D;
    float var  = s2 / (float)D - mean * mean;
    float inv  = rsqrtf(var + 1e-6f);
    if (obf16) {
        unsigned short* y = (unsigned short*)Yv + (size_t)row * Dp;
        for (int i = lane; i < Dp; i += 32)
            y[i] = (i < D) ? f2bf((x[i] - mean) * inv * g[i] + b[i]) : (unsigned short)0;
    } else {
        float* y = (float*)Yv + (size_t)row * D;
        for (int i = lane; i < D; i += 32)
            y[i] = (x[i] - mean) * inv * g[i] + b[i];
    }
}

// ---------------------------------------------------------------------------
// Pose MLP: one thread per edge (tiny: 17->48 LN lrelu 48->48 LN)
// ---------------------------------------------------------------------------
__global__ __launch_bounds__(64)
void pose_kernel(const float* __restrict__ ea, const float* __restrict__ w1,
                 const float* __restrict__ b1, const float* __restrict__ g1,
                 const float* __restrict__ bb1, const float* __restrict__ w2,
                 const float* __restrict__ b2, const float* __restrict__ g2,
                 const float* __restrict__ bb2, float* __restrict__ pose, int E)
{
    int e = blockIdx.x * blockDim.x + threadIdx.x;
    if (e >= E) return;
    float a[17];
    #pragma unroll
    for (int i = 0; i < 17; ++i) a[i] = ea[(size_t)e * 17 + i];

    float t[48];
    for (int j = 0; j < 48; ++j) {
        float s = b1[j];
        #pragma unroll
        for (int i = 0; i < 17; ++i) s += a[i] * w1[i * 48 + j];
        t[j] = s;
    }
    float m = 0.f, v2 = 0.f;
    for (int j = 0; j < 48; ++j) m += t[j];
    m *= (1.0f / 48.0f);
    for (int j = 0; j < 48; ++j) { float d = t[j] - m; v2 += d * d; }
    float inv = rsqrtf(v2 * (1.0f / 48.0f) + 1e-6f);
    for (int j = 0; j < 48; ++j) {
        float z = (t[j] - m) * inv * g1[j] + bb1[j];
        t[j] = (z > 0.f) ? z : 0.01f * z;
    }
    float u[48];
    for (int j = 0; j < 48; ++j) {
        float s = b2[j];
        for (int i = 0; i < 48; ++i) s += t[i] * w2[i * 48 + j];
        u[j] = s;
    }
    m = 0.f; v2 = 0.f;
    for (int j = 0; j < 48; ++j) m += u[j];
    m *= (1.0f / 48.0f);
    for (int j = 0; j < 48; ++j) { float d = u[j] - m; v2 += d * d; }
    inv = rsqrtf(v2 * (1.0f / 48.0f) + 1e-6f);
    for (int j = 0; j < 48; ++j)
        pose[(size_t)e * 48 + j] = (u[j] - m) * inv * g2[j] + bb2[j];
}

// ---------------------------------------------------------------------------
// Gather: act[e,s,c] = (s<32 ? x[dst[e],s,c] : x[src[e],s-32,c]) + pos[s,c]
//         for c<384, else pose[e,c-384]   (f32 out: layer0 residual stream)
// ---------------------------------------------------------------------------
__global__ __launch_bounds__(256)
void gather_kernel(const float* __restrict__ x, const int* __restrict__ dst,
                   const int* __restrict__ src, const float* __restrict__ pos,
                   const float* __restrict__ pose, float* __restrict__ act,
                   int eb, int ech)
{
    size_t i = (size_t)blockIdx.x * blockDim.x + threadIdx.x;
    size_t tot = (size_t)ech * 64 * 432;
    if (i >= tot) return;
    int c = (int)(i % 432);
    size_t t = i / 432;
    int s  = (int)(t % 64);
    int el = (int)(t / 64);
    int e  = eb + el;
    float v;
    if (c < 384) {
        int node = (s < 32) ? dst[e] : src[e];
        int ss   = (s < 32) ? s : (s - 32);
        v = x[((size_t)node * 32 + ss) * 384 + c] + pos[(size_t)s * 384 + c];
    } else {
        v = pose[(size_t)e * 48 + (c - 384)];
    }
    act[i] = v;
}

// ---------------------------------------------------------------------------
// Seq truncation repack (bf16, Dp-wide rows incl. zero pad):
//   (ech, sPrev, Dp) -> (ech, sNew, Dp)
// ---------------------------------------------------------------------------
__global__ __launch_bounds__(256)
void repack_kernel(const unsigned short* __restrict__ in,
                   unsigned short* __restrict__ out,
                   int ech, int sPrev, int sNew, int Dp)
{
    size_t i = (size_t)blockIdx.x * blockDim.x + threadIdx.x;
    size_t tot = (size_t)ech * sNew * Dp;
    if (i >= tot) return;
    int c = (int)(i % Dp);
    size_t t = i / Dp;
    int s  = (int)(t % sNew);
    int el = (int)(t / sNew);
    out[i] = in[((size_t)el * sPrev + s) * Dp + c];
}

// ---------------------------------------------------------------------------
// Softmax attention: one block per (edge,head), thread per query position.
// qkv f32 (dense 3D); out bf16 with row stride Dp; h==0 zeroes pad cols.
// ---------------------------------------------------------------------------
__global__ __launch_bounds__(64)
void attn_kernel(const float* __restrict__ qkv, unsigned short* __restrict__ out,
                 int ech, int S, int D, int Dp, int hd)
{
    int e = blockIdx.x;
    int h = blockIdx.y;
    int q = threadIdx.x;
    if (q >= S) return;
    const float* base = qkv + (size_t)e * S * 3 * D;
    const float scale = rsqrtf((float)hd);

    float qv[40];
    for (int i = 0; i < hd; ++i)
        qv[i] = base[(size_t)q * 3 * D + h * hd + i] * scale;

    float sc[64];
    float mx = -1e30f;
    for (int k = 0; k < S; ++k) {
        const float* kr = base + (size_t)k * 3 * D + D + h * hd;
        float s = 0.f;
        for (int i = 0; i < hd; ++i) s += qv[i] * kr[i];
        sc[k] = s;
        mx = fmaxf(mx, s);
    }
    float den = 0.f;
    for (int k = 0; k < S; ++k) { sc[k] = __expf(sc[k] - mx); den += sc[k]; }
    float idn = 1.0f / den;

    float o[40];
    for (int i = 0; i < hd; ++i) o[i] = 0.f;
    for (int k = 0; k < S; ++k) {
        const float* vr = base + (size_t)k * 3 * D + 2 * D + h * hd;
        float a = sc[k] * idn;
        for (int i = 0; i < hd; ++i) o[i] += a * vr[i];
    }
    unsigned short* orow = out + ((size_t)e * S + q) * Dp;
    unsigned short* oh   = orow + h * hd;
    for (int i = 0; i < hd; ++i) oh[i] = f2bf(o[i]);
    if (h == 0)
        for (int i = D; i < Dp; ++i) orow[i] = 0;   // zero K-pad columns
}

// ---------------------------------------------------------------------------
// segment_sum scatter: d_out[dst[e], s, c] += act[e, s, c]   (32 x 96)
// ---------------------------------------------------------------------------
__global__ __launch_bounds__(256)
void scatter_kernel(const float* __restrict__ act, const int* __restrict__ dst,
                    float* __restrict__ out, int eb, int ech)
{
    size_t i = (size_t)blockIdx.x * blockDim.x + threadIdx.x;
    size_t tot = (size_t)ech * 32 * 96;
    if (i >= tot) return;
    int rem = (int)(i % (32 * 96));
    int el  = (int)(i / (32 * 96));
    atomicAdd(&out[(size_t)dst[eb + el] * (32 * 96) + rem], act[i]);
}

__global__ __launch_bounds__(256)
void zero_kernel(float* __restrict__ p, size_t n)
{
    size_t i = (size_t)blockIdx.x * blockDim.x + threadIdx.x;
    if (i < n) p[i] = 0.f;
}

// ---------------------------------------------------------------------------
// Host orchestration
// ---------------------------------------------------------------------------
static inline void launch_gemm(const unsigned short* A, const unsigned short* Bt,
                               const float* bias, const float* resid, void* C,
                               int M, int Kp, int N, int act, int cbf16,
                               hipStream_t s)
{
    dim3 grid(M / BM, (N + BN - 1) / BN);
    gemm_bf16_wmma<<<grid, 256, 0, s>>>(A, Bt, bias, resid, C, M, Kp, N, act, cbf16);
}

static inline void launch_ln(const float* X, const float* g, const float* b,
                             void* Y, int M, int D, int Dp, int obf16, hipStream_t s)
{
    ln_kernel<<<(M + 3) / 4, 128, 0, s>>>(X, g, b, Y, M, D, Dp, obf16);
}

static inline void launch_wconv(const float* W, unsigned short* Wt, int K, int N,
                                hipStream_t s)
{
    int Kp = ceil32(K);
    size_t tot = (size_t)N * Kp;
    wconv_kernel<<<(unsigned)((tot + 255) / 256), 256, 0, s>>>(W, Wt, K, Kp, N);
}

struct LayerP {
    const float *in_w, *in_b, *ln1_g, *ln1_b, *qkv_w, *qkv_b, *proj_w, *proj_b,
                *ln2_g, *ln2_b, *fc1_w, *fc1_b, *fc2_w, *fc2_b, *oln_g, *oln_b;
};
struct LayerW {     // bf16 pre-transposed weights ([Np][Kp])
    unsigned short *in_wt, *qkv_wt, *proj_wt, *fc1_wt, *fc2_wt;
};

extern "C" void kernel_launch(void* const* d_in, const int* in_sizes, int n_in,
                              void* d_out, int out_size, void* d_ws, size_t ws_size,
                              hipStream_t stream)
{
    (void)in_sizes; (void)n_in; (void)out_size; (void)ws_size;

    const float* x     = (const float*)d_in[0];
    const int*   eidx  = (const int*)  d_in[1];     // [2, E]: row0=src, row1=dst
    const float* eattr = (const float*)d_in[2];
    const float* pos   = (const float*)d_in[3];
    const float* pm[8];
    for (int i = 0; i < 8; ++i) pm[i] = (const float*)d_in[4 + i];

    static const int DIMS[8] = {432, 384, 336, 288, 240, 192, 144, 96};
    static const int SEQS[8] = {64, 64, 56, 56, 48, 48, 40, 32};

    LayerP L[8];
    int p = 12;
    for (int l = 0; l < 8; ++l) {
        if (l > 0) { L[l].in_w = (const float*)d_in[p++]; L[l].in_b = (const float*)d_in[p++]; }
        else       { L[l].in_w = nullptr;                  L[l].in_b = nullptr; }
        L[l].ln1_g  = (const float*)d_in[p++]; L[l].ln1_b  = (const float*)d_in[p++];
        L[l].qkv_w  = (const float*)d_in[p++]; L[l].qkv_b  = (const float*)d_in[p++];
        L[l].proj_w = (const float*)d_in[p++]; L[l].proj_b = (const float*)d_in[p++];
        L[l].ln2_g  = (const float*)d_in[p++]; L[l].ln2_b  = (const float*)d_in[p++];
        L[l].fc1_w  = (const float*)d_in[p++]; L[l].fc1_b  = (const float*)d_in[p++];
        L[l].fc2_w  = (const float*)d_in[p++]; L[l].fc2_b  = (const float*)d_in[p++];
        L[l].oln_g  = (const float*)d_in[p++]; L[l].oln_b  = (const float*)d_in[p++];
    }

    // ---- workspace carve-out ----
    char*  ws  = (char*)d_ws;
    size_t off = 0;
    auto carve = [&](size_t bytes) -> char* {
        char* r = ws + off;
        off += (bytes + 255) & ~(size_t)255;
        return r;
    };
    const int DPMAX = 448;     // ceil32(432)
    float* POSE  = (float*)carve((size_t)E_TOT * 48 * 4);
    float* ACT   = (float*)carve((size_t)ECH * 64 * 432 * 4);          // residual (f32)
    float* ACT2  = (float*)carve((size_t)ECH * 64 * 432 * 4);          // in_proj out (f32)
    float* QKV   = (float*)carve((size_t)ECH * 64 * 3 * 432 * 4);      // f32
    unsigned short* BACT  = (unsigned short*)carve((size_t)ECH * 64 * DPMAX * 2); // out_ln bf16
    unsigned short* TMPH  = (unsigned short*)carve((size_t)ECH * 64 * DPMAX * 2); // repacked bf16
    unsigned short* LN1H  = (unsigned short*)carve((size_t)ECH * 64 * DPMAX * 2); // ln out bf16
    unsigned short* ATTNH = (unsigned short*)carve((size_t)ECH * 64 * DPMAX * 2); // attn out bf16
    unsigned short* HBUFH = (unsigned short*)carve((size_t)ECH * 64 * 1728 * 2);  // fc1 out bf16

    // bf16 transposed weights: [Np][Kp], Np rows carved for safe B-tile reads
    LayerW W[8];
    {
        int dPrev = 432;
        for (int l = 0; l < 8; ++l) {
            int d = DIMS[l];
            W[l].in_wt  = L[l].in_w
                ? (unsigned short*)carve((size_t)ceil64(d) * ceil32(dPrev) * 2) : nullptr;
            W[l].qkv_wt = (unsigned short*)carve((size_t)ceil64(3 * d) * ceil32(d) * 2);
            W[l].proj_wt= (unsigned short*)carve((size_t)ceil64(d) * ceil32(d) * 2);
            W[l].fc1_wt = (unsigned short*)carve((size_t)ceil64(4 * d) * ceil32(d) * 2);
            W[l].fc2_wt = (unsigned short*)carve((size_t)ceil64(d) * (size_t)(4 * d) * 2);
            dPrev = d;
        }
    }

    const int* srcIdx = eidx;          // edge_index[0]
    const int* dstIdx = eidx + E_TOT;  // edge_index[1]

    // ---- weight conversion (once per launch) ----
    {
        int dPrev = 432;
        for (int l = 0; l < 8; ++l) {
            int d = DIMS[l];
            if (L[l].in_w) launch_wconv(L[l].in_w, W[l].in_wt, dPrev, d, stream);
            launch_wconv(L[l].qkv_w,  W[l].qkv_wt,  d,     3 * d, stream);
            launch_wconv(L[l].proj_w, W[l].proj_wt, d,     d,     stream);
            launch_wconv(L[l].fc1_w,  W[l].fc1_wt,  d,     4 * d, stream);
            launch_wconv(L[l].fc2_w,  W[l].fc2_wt,  4 * d, d,     stream);
            dPrev = d;
        }
    }

    // ---- zero output accumulator ----
    {
        size_t osz = (size_t)NNODE * 32 * 96;
        zero_kernel<<<(unsigned)((osz + 255) / 256), 256, 0, stream>>>((float*)d_out, osz);
    }
    // ---- pose MLP (whole edge set) ----
    pose_kernel<<<(E_TOT + 63) / 64, 64, 0, stream>>>(
        eattr, pm[0], pm[1], pm[2], pm[3], pm[4], pm[5], pm[6], pm[7], POSE, E_TOT);

    for (int eb = 0; eb < E_TOT; eb += ECH) {
        {
            size_t tot = (size_t)ECH * 64 * 432;
            gather_kernel<<<(unsigned)((tot + 255) / 256), 256, 0, stream>>>(
                x, dstIdx, srcIdx, pos, POSE, ACT, eb, ECH);
        }
        int sPrev = 64, dPrev = 432;
        for (int l = 0; l < 8; ++l) {
            const int d  = DIMS[l], s = SEQS[l];
            const int Dp = ceil32(d), DpPrev = ceil32(dPrev);
            const int M  = ECH * s;
            const float* Xres = ACT;                     // residual stream (f32)
            if (l > 0) {
                const unsigned short* Ah = BACT;         // prev out_ln (bf16, DpPrev)
                if (s != sPrev) {
                    size_t tot = (size_t)ECH * s * DpPrev;
                    repack_kernel<<<(unsigned)((tot + 255) / 256), 256, 0, stream>>>(
                        BACT, TMPH, ECH, sPrev, s, DpPrev);
                    Ah = TMPH;
                }
                launch_gemm(Ah, W[l].in_wt, L[l].in_b, nullptr, ACT2,
                            M, DpPrev, d, 0, 0, stream);
                Xres = ACT2;
            }
            // attention sub-block
            launch_ln(Xres, L[l].ln1_g, L[l].ln1_b, LN1H, M, d, Dp, 1, stream);
            launch_gemm(LN1H, W[l].qkv_wt, L[l].qkv_b, nullptr, QKV,
                        M, Dp, 3 * d, 0, 0, stream);
            attn_kernel<<<dim3(ECH, HEADS), 64, 0, stream>>>(
                QKV, ATTNH, ECH, s, d, Dp, d / HEADS);
            launch_gemm(ATTNH, W[l].proj_wt, L[l].proj_b, Xres, ACT,
                        M, Dp, d, 0, 0, stream);
            // MLP sub-block
            launch_ln(ACT, L[l].ln2_g, L[l].ln2_b, LN1H, M, d, Dp, 1, stream);
            launch_gemm(LN1H, W[l].fc1_wt, L[l].fc1_b, nullptr, HBUFH,
                        M, Dp, 4 * d, 1, 1, stream);
            launch_gemm(HBUFH, W[l].fc2_wt, L[l].fc2_b, ACT, ACT,
                        M, 4 * d, d, 0, 0, stream);      // 4d is a multiple of 32
            // out LN: bf16 (feeds next layer) except last layer -> f32 for scatter
            if (l < 7) launch_ln(ACT, L[l].oln_g, L[l].oln_b, BACT, M, d, Dp, 1, stream);
            else       launch_ln(ACT, L[l].oln_g, L[l].oln_b, ACT,  M, d, Dp, 0, stream);
            sPrev = s; dPrev = d;
        }
        {
            size_t tot = (size_t)ECH * 32 * 96;
            scatter_kernel<<<(unsigned)((tot + 255) / 256), 256, 0, stream>>>(
                ACT, dstIdx, (float*)d_out, eb, ECH);
        }
    }
}
